// ImportranceRenderer_35854386987388
// MI455X (gfx1250) — matrix operand order repaired and checked by reading, not compiled, not run on hardware
//
#include <hip/hip_runtime.h>
#include <math.h>

// ---------------------------------------------------------------------------
// EG3D-style importance renderer for MI455X (gfx1250, wave32, WMMA).
// MLP (32->64->4) runs on v_wmma_f32_16x16x32_f16: 16 sample points per wave
// form the 16x32 A matrix; weights live in B-layout f16 registers; f32 accum.
// Bilinear tri-plane gathers stay fp32 (L2-resident: 100MB tensor vs 192MB L2).
// Transcendentals use the fast HW path (v_exp_f32/v_log_f32, TRANS32 units
// that co-execute with WMMA) instead of precise libm expansions.
// ---------------------------------------------------------------------------

typedef __attribute__((ext_vector_type(16))) _Float16 v16h;
typedef __attribute__((ext_vector_type(8)))  float    v8f;

#define BB     4
#define RR     2048
#define NSAMP  48                  // NS == NI == 48
#define NRAYS  (BB * RR)           // 8192
#define NPTS   (NRAYS * NSAMP)     // 393216 points per pass
#define NTILES (NPTS / 16)         // 24576 WMMA tiles per pass
#define CCH    32
#define HPX    256
#define WPX    256
#define PLANE_STRIDE (CCH * HPX * WPX)   // floats per plane (c-major)
#define RAY_S  2.25f
#define RAY_E  3.30f
#define DDELTA ((RAY_E - RAY_S) / (NSAMP - 1))

// workspace layout (float offsets)
#define OFF_DC 0                         // depths_coarse  [NPTS]
#define OFF_DF (OFF_DC + NPTS)           // depths_fine    [NPTS]
#define OFF_CC (OFF_DF + NPTS)           // colors_coarse  [NPTS*3]
#define OFF_SC (OFF_CC + NPTS * 3)       // dens_coarse    [NPTS]
#define OFF_CF (OFF_SC + NPTS)           // colors_fine    [NPTS*3]
#define OFF_SF (OFF_CF + NPTS * 3)       // dens_fine      [NPTS]
#define OFF_MM (OFF_SF + NPTS)           // minmax (2 uints)

__device__ __forceinline__ int   imin(int a, int b)   { return a < b ? a : b; }
__device__ __forceinline__ int   imax(int a, int b)   { return a > b ? a : b; }
// fast softplus/sigmoid on the HW transcendental pipe
__device__ __forceinline__ float softplusf(float x)
{
    return fmaxf(x, 0.f) + __logf(1.f + __expf(-fabsf(x)));
}
__device__ __forceinline__ float sigmoidf(float x)
{
    return __fdividef(1.f, 1.f + __expf(-x));
}

// ---------------------------------------------------------------------------
// Kernel 0: depths_coarse = linspace + noise*delta ; init minmax slots
// ---------------------------------------------------------------------------
__global__ __launch_bounds__(256)
void depths_init_kernel(const float* __restrict__ noise, float* __restrict__ ws)
{
    int gid = blockIdx.x * blockDim.x + threadIdx.x;
    if (gid == 0) {
        unsigned* mm = (unsigned*)(ws + OFF_MM);
        mm[0] = 0x7f800000u;   // +inf  (positive-float uint ordering)
        mm[1] = 0u;
    }
    if (gid < NPTS) {
        int s = gid % NSAMP;
        ws[OFF_DC + gid] = RAY_S + (float)s * DDELTA + noise[gid] * DDELTA;
    }
}

// ---------------------------------------------------------------------------
// Eval kernel: tri-plane bilinear gather + WMMA MLP. One wave = 16 points.
// A layout (16-bit 16x32): lane L holds row M=L&15; halves (v,p):
//   K = (v<4 ? 0 : 16) + 8*(L>>4) + 2*(v&3) + p
// B layout (16-bit 32x16): lane L holds col N=L&15; halves (v,p):
//   K = (v<4 ? 0 : 16) + 8*(L>>4) + 2*(v&3) + p
// C/D layout (f32 16x16): lane L holds (M = r + 8*(L>>4), N = L&15).
// ---------------------------------------------------------------------------
__global__ __launch_bounds__(256)
void eval_kernel(const float* __restrict__ features,
                 const float* __restrict__ rays_o,
                 const float* __restrict__ rays_d,
                 const float* __restrict__ w1,
                 const float* __restrict__ b1,
                 const float* __restrict__ w2,
                 const float* __restrict__ b2,
                 const float* __restrict__ depths,
                 float* __restrict__ colors,
                 float* __restrict__ dens)
{
    __shared__ _Float16 hbuf[8][16][64];     // per-wave 16x64 hidden tile

    const int lane = threadIdx.x & 31;
    const int wib  = threadIdx.x >> 5;
    const int g    = lane >> 4;              // half-wave group
    const int n    = lane & 15;              // column index / point-in-tile
    const int nwaves = (gridDim.x * blockDim.x) >> 5;
    const int gwave  = (blockIdx.x * blockDim.x + threadIdx.x) >> 5;

    // ---- load MLP weights once per wave into WMMA B-layout f16 registers ----
    v16h w1h[4];
    v8f  bias1[4];
#pragma unroll
    for (int cb = 0; cb < 4; ++cb) {
#pragma unroll
        for (int h = 0; h < 16; ++h) {
            int v = h >> 1, p = h & 1;
            int k = (v < 4) ? (8 * g + 2 * v + p) : (16 + 8 * g + 2 * (v - 4) + p);
            w1h[cb][h] = (_Float16)w1[k * 64 + cb * 16 + n];
        }
        float bv = b1[cb * 16 + n];
#pragma unroll
        for (int r = 0; r < 8; ++r) bias1[cb][r] = bv;
    }
    v16h w2h[2];
#pragma unroll
    for (int m = 0; m < 2; ++m)
#pragma unroll
        for (int h = 0; h < 16; ++h) {
            int v = h >> 1, p = h & 1;
            int k = 32 * m + ((v < 4) ? (8 * g + 2 * v + p) : (16 + 8 * g + 2 * (v - 4) + p));
            w2h[m][h] = (n < 4) ? (_Float16)w2[k * 4 + n] : (_Float16)0.0f;
        }
    v8f bias2;
    {
        float bv = (n < 4) ? b2[n] : 0.f;
#pragma unroll
        for (int r = 0; r < 8; ++r) bias2[r] = bv;
    }

    const int iters = (NTILES + nwaves - 1) / nwaves;
    for (int it = 0; it < iters; ++it) {
        const int  tile   = gwave + it * nwaves;   // wave-uniform
        const bool active = tile < NTILES;

        if (active) {
            // ---- this lane's point (lane L and L+16 share a point) ----
            const int pt  = tile * 16 + n;
            const int ray = pt / NSAMP;            // global ray = b*R + r
            const int b   = ray / RR;
            const float d = depths[pt];
            const float ox = rays_o[ray * 3 + 0], oy = rays_o[ray * 3 + 1], oz = rays_o[ray * 3 + 2];
            const float dx = rays_d[ray * 3 + 0], dy = rays_d[ray * 3 + 1], dz = rays_d[ray * 3 + 2];
            // coords = (2/BOX_WARP) * xyz ; plane projections (x,y):
            //   P0:(cx,cy)  P1:(cx,cz)  P2:(cz,cx)
            const float cx = 2.f * (ox + d * dx);
            const float cy = 2.f * (oy + d * dy);
            const float cz = 2.f * (oz + d * dz);
            const float gxa[3] = { cx, cx, cz };
            const float gya[3] = { cy, cz, cx };

            unsigned bases[12];
            float    wgts[12];
#pragma unroll
            for (int pl = 0; pl < 3; ++pl) {
                float x = (gxa[pl] + 1.f) * (WPX * 0.5f) - 0.5f;
                float y = (gya[pl] + 1.f) * (HPX * 0.5f) - 0.5f;
                float x0f = floorf(x), y0f = floorf(y);
                int   x0 = (int)x0f,  y0 = (int)y0f;
                float wx = x - x0f,   wy = y - y0f;
                unsigned pb = (unsigned)(b * 3 + pl) * PLANE_STRIDE;
#pragma unroll
                for (int k4 = 0; k4 < 4; ++k4) {
                    int ddx = k4 & 1, ddy = k4 >> 1;
                    int xi = x0 + ddx, yi = y0 + ddy;
                    float valid = (xi >= 0 && xi < WPX && yi >= 0 && yi < HPX) ? 1.f : 0.f;
                    int xc = imin(imax(xi, 0), WPX - 1);
                    int yc = imin(imax(yi, 0), HPX - 1);
                    wgts[pl * 4 + k4]  = (ddx ? wx : 1.f - wx) * (ddy ? wy : 1.f - wy) * valid;
                    bases[pl * 4 + k4] = pb + (unsigned)(yc * WPX + xc);
                }
            }

            // ---- gather 16 feature halves in WMMA A layout ----
            v16h a;
#pragma unroll
            for (int h = 0; h < 16; ++h) {
                int v = h >> 1, p = h & 1;
                int c = (v < 4) ? (8 * g + 2 * v + p) : (16 + 8 * g + 2 * (v - 4) + p);
                float acc = 0.f;
#pragma unroll
                for (int i = 0; i < 12; ++i)
                    acc += wgts[i] * features[(size_t)bases[i] + (size_t)c * (HPX * WPX)];
                a[h] = (_Float16)(acc * (1.f / 3.f));
            }

            // ---- layer 1: h = softplus(A(16x32) x W1(32x64) + b1) : 4 WMMAs ----
            v8f acc1[4];
#pragma unroll
            for (int cb = 0; cb < 4; ++cb)
                acc1[cb] = __builtin_amdgcn_wmma_f32_16x16x32_f16(
                    false, a, false, w1h[cb], (short)0, bias1[cb], false, false);

            // D layout -> row-major LDS tile (with softplus)
#pragma unroll
            for (int cb = 0; cb < 4; ++cb)
#pragma unroll
                for (int r = 0; r < 8; ++r)
                    hbuf[wib][r + 8 * g][cb * 16 + n] = (_Float16)softplusf(acc1[cb][r]);
        }
        __syncthreads();
        if (active) {
            // ---- reload hidden tile in A layout, layer 2: 2 WMMAs (K=64) ----
            v16h a2lo, a2hi;
#pragma unroll
            for (int h = 0; h < 16; ++h) {
                int v = h >> 1, p = h & 1;
                int k = (v < 4) ? (8 * g + 2 * v + p) : (16 + 8 * g + 2 * (v - 4) + p);
                a2lo[h] = hbuf[wib][n][k];
                a2hi[h] = hbuf[wib][n][32 + k];
            }
            v8f acc2 = __builtin_amdgcn_wmma_f32_16x16x32_f16(
                false, a2lo, false, w2h[0], (short)0, bias2, false, false);
            acc2 = __builtin_amdgcn_wmma_f32_16x16x32_f16(
                false, a2hi, false, w2h[1], (short)0, acc2, false, false);

            // ---- scatter: col 0 = sigma, cols 1..3 = sigmoid(rgb) ----
            const int ptbase = tile * 16;
            if (n == 0) {
#pragma unroll
                for (int r = 0; r < 8; ++r)
                    dens[ptbase + r + 8 * g] = acc2[r];
            } else if (n < 4) {
#pragma unroll
                for (int r = 0; r < 8; ++r)
                    colors[(ptbase + r + 8 * g) * 3 + (n - 1)] = sigmoidf(acc2[r]);
            }
        }
        __syncthreads();
    }
}

// ---------------------------------------------------------------------------
// Kernel B: per-ray coarse march -> importance PDF -> depths_fine
// ---------------------------------------------------------------------------
__global__ __launch_bounds__(128)
void march_pdf_kernel(const float* __restrict__ ws_dc,
                      const float* __restrict__ ws_sc,
                      const float* __restrict__ u_imp,
                      float* __restrict__ ws_df)
{
    int ray = blockIdx.x * blockDim.x + threadIdx.x;
    if (ray >= NRAYS) return;
    const float* dc = ws_dc + ray * NSAMP;
    const float* sc = ws_sc + ray * NSAMP;

    float w[NSAMP - 1], zmid[NSAMP - 1];
    float T = 1.f;
    for (int i = 0; i < NSAMP - 1; ++i) {
        float d0 = dc[i], d1 = dc[i + 1];
        zmid[i] = 0.5f * (d0 + d1);
        float sp    = softplusf(0.5f * (sc[i] + sc[i + 1]) - 1.f);
        float alpha = 1.f - __expf(-sp * (d1 - d0));
        w[i] = alpha * T;
        T *= (1.f - alpha + 1e-10f);
    }
    // pdf weights: wa[1..45] + eps with wa[i]=0.5*(wm[i]+wm[i+1])+0.01,
    // wm[j]=max(w[j-1],w[j]) interior  ->  pw[i]=0.5*(max(w[i],w[i+1])+max(w[i+1],w[i+2]))+0.01+1e-5
    float pw[45], cdf[46];
    float wsum = 0.f;
    for (int i = 0; i < 45; ++i) {
        float m1 = fmaxf(w[i], w[i + 1]);
        float m2 = fmaxf(w[i + 1], w[i + 2]);
        pw[i] = 0.5f * (m1 + m2) + 0.01f + 1e-5f;
        wsum += pw[i];
    }
    float inv = __fdividef(1.f, wsum);
    cdf[0] = 0.f;
    for (int i = 0; i < 45; ++i) cdf[i + 1] = cdf[i] + pw[i] * inv;

    for (int j = 0; j < NSAMP; ++j) {
        float u = u_imp[ray * NSAMP + j];
        int lo = 0, hi = 46;                      // searchsorted 'right'
        while (lo < hi) { int mid = (lo + hi) >> 1; if (cdf[mid] <= u) lo = mid + 1; else hi = mid; }
        int below = imax(lo - 1, 0);
        int above = imin(lo, 45);
        float c0 = cdf[below], c1 = cdf[above];
        float denom = c1 - c0;
        if (denom < 1e-5f) denom = 1.f;
        float b0v = zmid[below], b1v = zmid[above];
        ws_df[ray * NSAMP + j] = b0v + (u - c0) / denom * (b1v - b0v);
    }
}

// ---------------------------------------------------------------------------
// Minmax over all depths (coarse+fine contiguous) for the depth clip.
// Positive floats -> uint ordering, so uint atomics are exact.
// ---------------------------------------------------------------------------
__global__ __launch_bounds__(256)
void minmax_kernel(const float* __restrict__ d, int nelem, unsigned* __restrict__ mm)
{
    int tid    = blockIdx.x * blockDim.x + threadIdx.x;
    int stride = gridDim.x * blockDim.x;
    float mn = INFINITY, mx = -INFINITY;
    for (int i = tid; i < nelem; i += stride) {
        float v = d[i];
        mn = fminf(mn, v);
        mx = fmaxf(mx, v);
    }
#pragma unroll
    for (int off = 16; off > 0; off >>= 1) {
        mn = fminf(mn, __shfl_down(mn, off, 32));
        mx = fmaxf(mx, __shfl_down(mx, off, 32));
    }
    if ((threadIdx.x & 31) == 0) {
        atomicMin(&mm[0], __float_as_uint(mn));
        atomicMax(&mm[1], __float_as_uint(mx));
    }
}

// ---------------------------------------------------------------------------
// Kernel D: stable sort fine depths, merge with (already sorted) coarse,
// final ray march, write rgb / depth / weight-sum.
// ---------------------------------------------------------------------------
__global__ __launch_bounds__(128)
void final_kernel(const float* __restrict__ ws, float* __restrict__ out)
{
    int ray = blockIdx.x * blockDim.x + threadIdx.x;
    if (ray >= NRAYS) return;

    const float* dc  = ws + OFF_DC + ray * NSAMP;
    const float* df  = ws + OFF_DF + ray * NSAMP;
    const float* cc  = ws + OFF_CC + ray * NSAMP * 3;
    const float* scd = ws + OFF_SC + ray * NSAMP;
    const float* cf  = ws + OFF_CF + ray * NSAMP * 3;
    const float* sfd = ws + OFF_SF + ray * NSAMP;
    const unsigned* mm = (const unsigned*)(ws + OFF_MM);
    const float dmin = __uint_as_float(mm[0]);
    const float dmax = __uint_as_float(mm[1]);

    // stable insertion sort of the 48 fine samples (coarse already sorted)
    float fd[NSAMP];
    unsigned char fo[NSAMP];
    for (int i = 0; i < NSAMP; ++i) { fd[i] = df[i]; fo[i] = (unsigned char)i; }
    for (int i = 1; i < NSAMP; ++i) {
        float kd = fd[i]; unsigned char ki = fo[i];
        int j = i - 1;
        while (j >= 0 && fd[j] > kd) { fd[j + 1] = fd[j]; fo[j + 1] = fo[j]; --j; }
        fd[j + 1] = kd; fo[j + 1] = ki;
    }

    // merge-march over the 96 sorted samples (coarse wins ties == stable argsort)
    int ic = 0, ifn = 0;
    float pd = 0.f, ps = 0.f, pc0 = 0.f, pc1 = 0.f, pc2 = 0.f;
    float T = 1.f, rgb0 = 0.f, rgb1 = 0.f, rgb2 = 0.f, dep = 0.f, wtot = 0.f;
    for (int step = 0; step < 2 * NSAMP; ++step) {
        bool takeC = (ic < NSAMP) && (ifn >= NSAMP || dc[ic] <= fd[ifn]);
        float nd, ns, nc0, nc1, nc2;
        if (takeC) {
            nd = dc[ic]; ns = scd[ic];
            nc0 = cc[ic * 3 + 0]; nc1 = cc[ic * 3 + 1]; nc2 = cc[ic * 3 + 2];
            ++ic;
        } else {
            int src = fo[ifn];
            nd = fd[ifn]; ns = sfd[src];
            nc0 = cf[src * 3 + 0]; nc1 = cf[src * 3 + 1]; nc2 = cf[src * 3 + 2];
            ++ifn;
        }
        if (step > 0) {
            float delta = nd - pd;
            float sp    = softplusf(0.5f * (ps + ns) - 1.f);
            float alpha = 1.f - __expf(-sp * delta);
            float wgt   = alpha * T;
            T *= (1.f - alpha + 1e-10f);
            rgb0 += wgt * 0.5f * (pc0 + nc0);
            rgb1 += wgt * 0.5f * (pc1 + nc1);
            rgb2 += wgt * 0.5f * (pc2 + nc2);
            dep  += wgt * 0.5f * (pd + nd);
            wtot += wgt;
        }
        pd = nd; ps = ns; pc0 = nc0; pc1 = nc1; pc2 = nc2;
    }

    float cd = dep / wtot;
    if (!(cd == cd)) cd = INFINITY;           // nan_to_num(nan=inf)
    cd = fminf(fmaxf(cd, dmin), dmax);

    out[ray * 3 + 0] = rgb0 * 2.f - 1.f;      // rgb_final (B,R,3)
    out[ray * 3 + 1] = rgb1 * 2.f - 1.f;
    out[ray * 3 + 2] = rgb2 * 2.f - 1.f;
    out[NRAYS * 3 + ray]          = cd;       // depth_final (B,R,1)
    out[NRAYS * 3 + NRAYS + ray]  = wtot;     // weight sum  (B,R,1)
}

// ---------------------------------------------------------------------------
extern "C" void kernel_launch(void* const* d_in, const int* in_sizes, int n_in,
                              void* d_out, int out_size, void* d_ws, size_t ws_size,
                              hipStream_t stream)
{
    const float* features = (const float*)d_in[0];
    const float* rays_o   = (const float*)d_in[1];
    const float* rays_d   = (const float*)d_in[2];
    const float* w1       = (const float*)d_in[3];
    const float* b1       = (const float*)d_in[4];
    const float* w2       = (const float*)d_in[5];
    const float* b2       = (const float*)d_in[6];
    const float* noise    = (const float*)d_in[7];
    const float* u_imp    = (const float*)d_in[8];
    float* out = (float*)d_out;
    float* ws  = (float*)d_ws;

    // 0) coarse depths + minmax init
    depths_init_kernel<<<(NPTS + 255) / 256, 256, 0, stream>>>(noise, ws);

    // 1) coarse eval (WMMA MLP)
    eval_kernel<<<256, 256, 0, stream>>>(features, rays_o, rays_d, w1, b1, w2, b2,
                                         ws + OFF_DC, ws + OFF_CC, ws + OFF_SC);

    // 2) coarse march + PDF inversion -> fine depths
    march_pdf_kernel<<<(NRAYS + 127) / 128, 128, 0, stream>>>(
        ws + OFF_DC, ws + OFF_SC, u_imp, ws + OFF_DF);

    // 3) fine eval (WMMA MLP)
    eval_kernel<<<256, 256, 0, stream>>>(features, rays_o, rays_d, w1, b1, w2, b2,
                                         ws + OFF_DF, ws + OFF_CF, ws + OFF_SF);

    // 4) global min/max over all depths (coarse+fine are contiguous)
    minmax_kernel<<<256, 256, 0, stream>>>(ws + OFF_DC, 2 * NPTS,
                                           (unsigned*)(ws + OFF_MM));

    // 5) sort + merge + final march -> outputs
    final_kernel<<<(NRAYS + 127) / 128, 128, 0, stream>>>(ws, out);
}
